// GNNRecommender_89146341196361
// MI455X (gfx1250) — matrix operand order repaired and checked by reading, not compiled, not run on hardware
//
#include <hip/hip_runtime.h>
#include <hip/hip_bf16.h>

#define ALPHA_TP 0.15f
#define K_ITER   10
#define LN_EPS   1e-5f
#define N_USER   50000
#define N_ITEM   100000
#define N_EDGE   1600000
#define N_BATCH  8192
// D_IN = 128, HID = 128, OUT = 64

typedef __attribute__((ext_vector_type(16))) __bf16 v16bf;
typedef __attribute__((ext_vector_type(8)))  __bf16 v8bf;
typedef __attribute__((ext_vector_type(8)))  float  v8f;

// ---------------------------------------------------------------------------
// 16x16 output tile = A(16x128, LDS bf16, row-major stride 128)
//                   x W^T(rows = output col n, 128 contiguous K, bf16)
// 4 chained v_wmma_f32_16x16x32_bf16, f32 accumulate.
// A-fragment lane layout (16-bit A, 16x32):
//   lanes 0-15  hold row M=lane,    K = {k0..k0+7, k0+16..k0+23}
//   lanes 16-31 hold row M=lane-16, K = {k0+8..k0+15, k0+24..k0+31}
// B-fragment: same K split as A (lanes 0-15 col N=lane K={k0..k0+7, k0+16..k0+23};
// lanes 16-31 K={k0+8..k0+15, k0+24..k0+31}).
// Both halves are 16B-contiguous -> b128 loads.
// ---------------------------------------------------------------------------
__device__ __forceinline__ v8f tile_mm128(const __bf16* __restrict__ As,
                                          const __bf16* __restrict__ Wt,
                                          int n0, int lr, int lhi) {
  v8f acc = {0.f, 0.f, 0.f, 0.f, 0.f, 0.f, 0.f, 0.f};
#pragma unroll
  for (int k0 = 0; k0 < 128; k0 += 32) {
    const __bf16* pa = As + lr * 128 + k0 + lhi * 8;
    v8bf alo = *(const v8bf*)(pa);
    v8bf ahi = *(const v8bf*)(pa + 16);
    v16bf a = __builtin_shufflevector(alo, ahi, 0, 1, 2, 3, 4, 5, 6, 7,
                                      8, 9, 10, 11, 12, 13, 14, 15);
    const __bf16* pb = Wt + (n0 + lr) * 128 + k0 + lhi * 8;
    v8bf blo = *(const v8bf*)(pb);
    v8bf bhi = *(const v8bf*)(pb + 16);
    v16bf b = __builtin_shufflevector(blo, bhi, 0, 1, 2, 3, 4, 5, 6, 7,
                                      8, 9, 10, 11, 12, 13, 14, 15);
    acc = __builtin_amdgcn_wmma_f32_16x16x32_bf16(false, a, false, b,
                                                  (short)0, acc, false, false);
  }
  return acc;
}

// Fused 3-layer encoder MLP: x[N,128] -> LN/ELU -> LN/ELU -> h0[N,64].
// One block = 16 rows, 8 wave32. Waves each own a 16-wide N-tile.
__global__ __launch_bounds__(256) void mlp3_kernel(
    const float* __restrict__ X, const __bf16* __restrict__ Wt,
    const float* __restrict__ b1, const float* __restrict__ g1, const float* __restrict__ be1,
    const float* __restrict__ b2, const float* __restrict__ g2, const float* __restrict__ be2,
    const float* __restrict__ b3, float* __restrict__ h0, float* __restrict__ h) {
  __shared__ __attribute__((aligned(16))) __bf16 As[16 * 128];
  __shared__ float Hs[16 * 128];
  __shared__ float mstat[16];
  __shared__ float vstat[16];

  const int tid  = threadIdx.x;
  const int wave = tid >> 5;
  const int lane = tid & 31;
  const int lr   = lane & 15;
  const int lhi  = lane >> 4;
  const long long row0 = (long long)blockIdx.x * 16;

  for (int i = tid; i < 16 * 128; i += 256) {
    int r = i >> 7, c = i & 127;
    As[i] = (__bf16)X[(row0 + r) * 128 + c];
  }
  __syncthreads();

#pragma unroll
  for (int layer = 0; layer < 2; ++layer) {
    const __bf16* W    = Wt + layer * (128 * 128);
    const float*  bias = layer ? b2 : b1;
    const float*  gg   = layer ? g2 : g1;
    const float*  bb   = layer ? be2 : be1;
    const int n0 = wave * 16;
    v8f acc = tile_mm128(As, W, n0, lr, lhi);
#pragma unroll
    for (int v = 0; v < 8; ++v)
      Hs[(v + lhi * 8) * 128 + n0 + lr] = acc[v] + bias[n0 + lr];
    __syncthreads();
    if (tid < 16) {  // per-row mean / population variance (matches jnp.var)
      float s = 0.f, s2 = 0.f;
      for (int c = 0; c < 128; ++c) {
        float x = Hs[tid * 128 + c];
        s += x; s2 += x * x;
      }
      float m = s * (1.f / 128.f);
      mstat[tid] = m;
      vstat[tid] = s2 * (1.f / 128.f) - m * m;
    }
    __syncthreads();
    for (int i = tid; i < 16 * 128; i += 256) {
      int r = i >> 7, c = i & 127;
      float x = (Hs[i] - mstat[r]) * rsqrtf(vstat[r] + LN_EPS) * gg[c] + bb[c];
      x = x > 0.f ? x : (expf(x) - 1.f);  // ELU(alpha=1)
      As[i] = (__bf16)x;
    }
    __syncthreads();
  }

  // Layer 3: 16x64 output, waves 0..3 (uniform per-wave branch, EXEC all-1s).
  if (wave < 4) {
    const int n0 = wave * 16;
    v8f acc = tile_mm128(As, Wt + 2 * 128 * 128, n0, lr, lhi);
#pragma unroll
    for (int v = 0; v < 8; ++v) {
      int m = v + lhi * 8;
      float val = acc[v] + b3[n0 + lr];
      long long o = (row0 + m) * 64 + n0 + lr;
      h0[o] = val;  // teleport anchor
      h[o]  = val;  // propagated state (initialized to h0)
    }
  }
}

// float32 [K,N] row-major -> bf16 [N,K] (transposed, contiguous K per row)
__global__ void cvt_w_kernel(const float* __restrict__ W, __bf16* __restrict__ Wt,
                             int K, int N) {
  int i = blockIdx.x * blockDim.x + threadIdx.x;
  if (i >= K * N) return;
  int k = i / N, n = i % N;
  Wt[n * K + k] = (__bf16)W[i];
}

__global__ void fill0_kernel(float* __restrict__ p, long long n) {
  long long i = (long long)blockIdx.x * blockDim.x + threadIdx.x;
  if (i < n) p[i] = 0.f;
}

__global__ void degree_kernel(const int* __restrict__ dst, float* __restrict__ deg, int E) {
  int e = blockIdx.x * blockDim.x + threadIdx.x;
  if (e < E) atomicAdd(&deg[dst[e]], 1.0f);
}

// Edge-parallel gather + scatter-add: 16 threads per edge, float4 per thread.
__global__ void scatter_kernel(const int* __restrict__ src, const int* __restrict__ dst,
                               const float* __restrict__ h, float* __restrict__ agg, int E) {
  long long t = (long long)blockIdx.x * blockDim.x + threadIdx.x;
  long long e = t >> 4;
  if (e >= E) return;
  int c = (int)(t & 15) * 4;
  long long s = src[e];
  long long d = dst[e];
  __builtin_prefetch(h + s * 64 + c, 0, 1);  // global_prefetch_b8
  const float4 v = *(const float4*)(h + s * 64 + c);
  float* base = agg + d * 64 + c;
  atomicAdd(base + 0, v.x);
  atomicAdd(base + 1, v.y);
  atomicAdd(base + 2, v.z);
  atomicAdd(base + 3, v.w);
}

// h = (1-alpha) * agg/deg + alpha * h0
__global__ void combine_kernel(float* __restrict__ h, const float* __restrict__ agg,
                               const float* __restrict__ h0, const float* __restrict__ deg,
                               long long n) {
  long long i = (long long)blockIdx.x * blockDim.x + threadIdx.x;
  if (i >= n) return;
  long long row = i >> 6;
  float dg = fmaxf(deg[row], 1.0f);
  h[i] = (1.0f - ALPHA_TP) * (agg[i] / dg) + ALPHA_TP * h0[i];
}

// One block (128 threads) per (pair, pos/neg) prediction.
__global__ __launch_bounds__(128) void predict_kernel(
    const float* __restrict__ hu, const float* __restrict__ hi,
    const int* __restrict__ uidx, const int* __restrict__ pidx, const int* __restrict__ nidx,
    const float* __restrict__ W1, const float* __restrict__ b1,
    const float* __restrict__ W2, const float* __restrict__ b2,
    const float* __restrict__ W3, const float* __restrict__ b3,
    float* __restrict__ scores, int B) {
  __shared__ float xin[128];
  __shared__ float h1[128];
  __shared__ float h2[64];
  int p = blockIdx.x;
  int b = p % B;
  int isneg = p / B;
  int t = threadIdx.x;
  long long u = uidx[b];
  long long it = isneg ? nidx[b] : pidx[b];
  xin[t] = (t < 64) ? hu[u * 64 + t] : hi[it * 64 + (t - 64)];
  __syncthreads();
  float a1 = b1[t];
  for (int k = 0; k < 128; ++k) a1 = fmaf(xin[k], W1[k * 128 + t], a1);
  h1[t] = fmaxf(a1, 0.f);
  __syncthreads();
  if (t < 64) {
    float a2 = b2[t];
    for (int k = 0; k < 128; ++k) a2 = fmaf(h1[k], W2[k * 64 + t], a2);
    h2[t] = fmaxf(a2, 0.f);
  }
  __syncthreads();
  if (t == 0) {
    float s = b3[0];
    for (int k = 0; k < 64; ++k) s = fmaf(h2[k], W3[k], s);
    scores[p] = s;
  }
}

// loss = mean_b softplus(-(pos_b - neg_b)), stable form.
__global__ __launch_bounds__(256) void loss_kernel(const float* __restrict__ scores,
                                                   float* __restrict__ out, int B) {
  __shared__ float red[256];
  int t = threadIdx.x;
  float acc = 0.f;
  for (int b = t; b < B; b += 256) {
    float s = scores[b] - scores[b + B];
    float z = -s;
    acc += fmaxf(z, 0.f) + log1pf(expf(-fabsf(z)));
  }
  red[t] = acc;
  __syncthreads();
  for (int off = 128; off > 0; off >>= 1) {
    if (t < off) red[t] += red[t + off];
    __syncthreads();
  }
  if (t == 0) out[0] = red[0] / (float)B;
}

extern "C" void kernel_launch(void* const* d_in, const int* in_sizes, int n_in,
                              void* d_out, int out_size, void* d_ws, size_t ws_size,
                              hipStream_t stream) {
  (void)in_sizes; (void)n_in; (void)out_size; (void)ws_size;
  const float* x_user   = (const float*)d_in[0];
  const float* x_item   = (const float*)d_in[1];
  const int*   edge_ui  = (const int*)d_in[2];   // [2, E]
  const int*   edge_iu  = (const int*)d_in[3];   // [2, E]
  const int*   user_idx = (const int*)d_in[4];
  const int*   pos_idx  = (const int*)d_in[5];
  const int*   neg_idx  = (const int*)d_in[6];
  // user-MLP params (7..16): W1,b1,g1,beta1,W2,b2,g2,beta2,W3,b3
  const float* uW1 = (const float*)d_in[7];
  const float* ub1 = (const float*)d_in[8];
  const float* ug1 = (const float*)d_in[9];
  const float* ue1 = (const float*)d_in[10];
  const float* uW2 = (const float*)d_in[11];
  const float* ub2 = (const float*)d_in[12];
  const float* ug2 = (const float*)d_in[13];
  const float* ue2 = (const float*)d_in[14];
  const float* uW3 = (const float*)d_in[15];
  const float* ub3 = (const float*)d_in[16];
  // item-MLP params (17..26)
  const float* iW1 = (const float*)d_in[17];
  const float* ib1 = (const float*)d_in[18];
  const float* ig1 = (const float*)d_in[19];
  const float* ie1 = (const float*)d_in[20];
  const float* iW2 = (const float*)d_in[21];
  const float* ib2 = (const float*)d_in[22];
  const float* ig2 = (const float*)d_in[23];
  const float* ie2 = (const float*)d_in[24];
  const float* iW3 = (const float*)d_in[25];
  const float* ib3 = (const float*)d_in[26];
  // predictor params (27..32)
  const float* pW1 = (const float*)d_in[27];
  const float* pb1 = (const float*)d_in[28];
  const float* pW2 = (const float*)d_in[29];
  const float* pb2 = (const float*)d_in[30];
  const float* pW3 = (const float*)d_in[31];
  const float* pb3 = (const float*)d_in[32];

  // ---- workspace carve (256B-aligned bumps) ----
  char* ws = (char*)d_ws;
  size_t off = 0;
  auto alloc = [&](size_t bytes) -> void* {
    void* p = ws + off;
    off += (bytes + 255) & ~(size_t)255;
    return p;
  };
  const int WELTS = 128 * 128 + 128 * 128 + 128 * 64;  // 40960 bf16 per MLP
  __bf16* wu   = (__bf16*)alloc((size_t)WELTS * 2);
  __bf16* wi   = (__bf16*)alloc((size_t)WELTS * 2);
  float* h0u   = (float*)alloc((size_t)N_USER * 64 * 4);
  float* h0i   = (float*)alloc((size_t)N_ITEM * 64 * 4);
  float* hu    = (float*)alloc((size_t)N_USER * 64 * 4);
  float* hi    = (float*)alloc((size_t)N_ITEM * 64 * 4);
  float* aggu  = (float*)alloc((size_t)N_USER * 64 * 4);
  float* aggi  = (float*)alloc((size_t)N_ITEM * 64 * 4);
  float* degu  = (float*)alloc((size_t)N_USER * 4);
  float* degi  = (float*)alloc((size_t)N_ITEM * 4);
  float* score = (float*)alloc((size_t)2 * N_BATCH * 4);

  const int T = 256;
  // weight conversion (f32 row-major -> bf16 transposed)
  cvt_w_kernel<<<(16384 + T - 1) / T, T, 0, stream>>>(uW1, wu,          128, 128);
  cvt_w_kernel<<<(16384 + T - 1) / T, T, 0, stream>>>(uW2, wu + 16384,  128, 128);
  cvt_w_kernel<<<(8192  + T - 1) / T, T, 0, stream>>>(uW3, wu + 32768,  128, 64);
  cvt_w_kernel<<<(16384 + T - 1) / T, T, 0, stream>>>(iW1, wi,          128, 128);
  cvt_w_kernel<<<(16384 + T - 1) / T, T, 0, stream>>>(iW2, wi + 16384,  128, 128);
  cvt_w_kernel<<<(8192  + T - 1) / T, T, 0, stream>>>(iW3, wi + 32768,  128, 64);

  // degrees
  fill0_kernel<<<(N_USER + T - 1) / T, T, 0, stream>>>(degu, N_USER);
  fill0_kernel<<<(N_ITEM + T - 1) / T, T, 0, stream>>>(degi, N_ITEM);
  degree_kernel<<<(N_EDGE + T - 1) / T, T, 0, stream>>>(edge_ui + N_EDGE, degi, N_EDGE);
  degree_kernel<<<(N_EDGE + T - 1) / T, T, 0, stream>>>(edge_iu + N_EDGE, degu, N_EDGE);

  // encoder MLPs (WMMA) — NU and NI are multiples of 16
  mlp3_kernel<<<N_USER / 16, 256, 0, stream>>>(x_user, wu, ub1, ug1, ue1,
                                               ub2, ug2, ue2, ub3, h0u, hu);
  mlp3_kernel<<<N_ITEM / 16, 256, 0, stream>>>(x_item, wi, ib1, ig1, ie1,
                                               ib2, ig2, ie2, ib3, h0i, hi);

  // APPNP propagation
  const long long nu64 = (long long)N_USER * 64;
  const long long ni64 = (long long)N_ITEM * 64;
  const long long scat_threads = (long long)N_EDGE * 16;
  for (int k = 0; k < K_ITER; ++k) {
    fill0_kernel<<<(unsigned)((ni64 + T - 1) / T), T, 0, stream>>>(aggi, ni64);
    fill0_kernel<<<(unsigned)((nu64 + T - 1) / T), T, 0, stream>>>(aggu, nu64);
    scatter_kernel<<<(unsigned)((scat_threads + T - 1) / T), T, 0, stream>>>(
        edge_ui, edge_ui + N_EDGE, hu, aggi, N_EDGE);
    scatter_kernel<<<(unsigned)((scat_threads + T - 1) / T), T, 0, stream>>>(
        edge_iu, edge_iu + N_EDGE, hi, aggu, N_EDGE);
    combine_kernel<<<(unsigned)((ni64 + T - 1) / T), T, 0, stream>>>(hi, aggi, h0i, degi, ni64);
    combine_kernel<<<(unsigned)((nu64 + T - 1) / T), T, 0, stream>>>(hu, aggu, h0u, degu, nu64);
  }

  // predictor + BPR loss
  predict_kernel<<<2 * N_BATCH, 128, 0, stream>>>(hu, hi, user_idx, pos_idx, neg_idx,
                                                  pW1, pb1, pW2, pb2, pW3, pb3,
                                                  score, N_BATCH);
  loss_kernel<<<1, 256, 0, stream>>>(score, (float*)d_out, N_BATCH);
}
